// MoEPredictor_90726889160863
// MI455X (gfx1250) — matrix operand order repaired and checked by reading, not compiled, not run on hardware
//
#include <hip/hip_runtime.h>
#include <math.h>

typedef _Float16 half_t;
typedef __attribute__((ext_vector_type(16))) _Float16 v16h;
typedef __attribute__((ext_vector_type(8)))  _Float16 v8h;
typedef __attribute__((ext_vector_type(8)))  float    v8f;

#define NEXP 6
#define OUTW 120          // 2*FUT
#define TM   128          // tokens per block
#define NTOK 49152        // 8192*6

// ---- swizzled f16 weight offsets inside workspace (in half elements) ----
static constexpr size_t OFS_RW1 = 0;                       // 128x256  -> 4*16*512
static constexpr size_t OFS_RW2 = OFS_RW1 + 32768;         // 256x128  -> 8*8*512
static constexpr size_t OFS_RW3 = OFS_RW2 + 32768;         // 128x16   -> 4*1*512
static constexpr size_t OFS_TW1 = OFS_RW3 + 2048;          // 6x 128x256
static constexpr size_t OFS_TW2 = OFS_TW1 + 6 * 32768;     // 6x 256x256
static constexpr size_t OFS_TW3 = OFS_TW2 + 6 * 65536;     // 6x 256x128(pad 120)
static constexpr size_t OFS_SW1 = OFS_TW3 + 6 * 32768;     // 6x 128x128
static constexpr size_t OFS_SW2 = OFS_SW1 + 6 * 16384;     // 6x 128x64
static constexpr size_t OFS_SW3 = OFS_SW2 + 6 * 8192;      // 6x 64x16(pad 1)

// Branch-free exact-erf GELU: erf via Abramowitz-Stegun 7.1.28, |err| <= 3e-7.
__device__ __forceinline__ float gelu_fast(float x) {
    float u = x * 0.7071067811865476f;
    float s = fabsf(u);
    float t = fmaf(s, 4.30638e-5f, 2.765672e-4f);   // a6*s + a5
    t = fmaf(s, t, 1.520143e-4f);                   // + a4
    t = fmaf(s, t, 9.2705272e-3f);                  // + a3
    t = fmaf(s, t, 4.22820123e-2f);                 // + a2
    t = fmaf(s, t, 7.05230784e-2f);                 // + a1
    t = fmaf(s, t, 1.0f);                           // + 1
    float t2 = t * t;
    float t4 = t2 * t2;
    float t8 = t4 * t4;
    float t16 = t8 * t8;
    float r = __builtin_amdgcn_rcpf(t16);           // (1+...)^-16
    float erfv = copysignf(1.0f - r, u);
    return 0.5f * x * (1.0f + erfv);
}

union AFrag { v16h v; v8h h[2]; };

// A fragment (16x32 f16, MxK): lane l holds row M=l&15, K = ((e<8)?0:16) + (l>>4)*8 + (e&7)
__device__ __forceinline__ v16h load_a(const half_t* A, int lda, int mbase, int kbase, int lane) {
    int row = mbase + (lane & 15);
    int off = (lane >> 4) << 3;
    const half_t* p = A + (size_t)row * lda + kbase + off;
    AFrag f;
    f.h[0] = *(const v8h*)(p);
    f.h[1] = *(const v8h*)(p + 16);
    return f.v;
}

// B fragment from pre-swizzled tile: lane-contiguous 16 halves
__device__ __forceinline__ v16h load_b(const half_t* Bsw, int lane) {
    return *(const v16h*)(Bsw + lane * 16);
}

__device__ __forceinline__ v8f wmma_f16(v16h a, v16h b, v8f c) {
    return __builtin_amdgcn_wmma_f32_16x16x32_f16(false, a, false, b, (short)0, c, false, false);
}

// Column-owning GEMM: each wave preloads its B column fragments into registers
// (reused across all m-tiles -> 8x less L2 weight traffic), sweeps m-tiles
// (unrolled x2 so two independent WMMA accumulation chains interleave on the
// XDL pipe), applies bias(+GELU), stores f16 to LDS. Narrow layers split m.
template<int Mt, int Nt, int Kt, bool GELU>
__device__ __forceinline__ void gemm_lds(const half_t* A, int lda, const half_t* Bsw,
                                         const float* bias, int biasN,
                                         half_t* C, int ldc, int wave, int lane) {
    constexpr int groups = (Nt >= 8) ? 1 : (8 / Nt);
    constexpr int mchunk = Mt / groups;
    const int g  = (Nt >= 8) ? 0 : (wave / Nt);
    const int n0 = (Nt >= 8) ? wave : (wave % Nt);
    for (int n = n0; n < Nt; n += 8) {
        v16h breg[Kt];
#pragma unroll
        for (int k = 0; k < Kt; ++k)
            breg[k] = load_b(Bsw + (size_t)(k * Nt + n) * 512, lane);
        int col = n * 16 + (lane & 15);
        float bv = (col < biasN) ? bias[col] : 0.0f;
#pragma unroll 2
        for (int m = g * mchunk; m < g * mchunk + mchunk; ++m) {
            v8f acc = {};
#pragma unroll
            for (int k = 0; k < Kt; ++k) {
                v16h a = load_a(A, lda, m * 16, k * 32, lane);
                acc = wmma_f16(a, breg[k], acc);
            }
            int mrow = m * 16 + ((lane >> 4) << 3);
#pragma unroll
            for (int v = 0; v < 8; ++v) {
                float xv = acc[v] + bv;
                if (GELU) xv = gelu_fast(xv);
                C[(size_t)(mrow + v) * ldc + col] = (half_t)xv;
            }
        }
    }
}

// Pack W[K][Nsrc] (row-major fp32, nmat stacked) into WMMA-B swizzled f16 with N padded to Npad
__global__ void pack_b_kernel(const float* __restrict__ W, half_t* __restrict__ out,
                              int K, int Nsrc, int Npad, long total) {
    long idx = (long)blockIdx.x * blockDim.x + threadIdx.x;
    if (idx >= total) return;
    int ntiles = Npad >> 4;
    long per = (long)(K >> 5) * ntiles * 512;
    long mat = idx / per;
    long r = idx - mat * per;
    int e    = (int)(r & 15);
    int lane = (int)((r >> 4) & 31);
    int tile = (int)(r >> 9);
    int nt = tile % ntiles;
    int kt = tile / ntiles;
    int k = kt * 32 + ((e < 8) ? 0 : 16) + ((lane >> 4) << 3) + (e & 7);   // B layout: K = ((e<8)?0:16) + (lane>>4)*8 + (e&7)
    int n = nt * 16 + (lane & 15);
    float v = (n < Nsrc) ? W[mat * (long)K * Nsrc + (long)k * Nsrc + n] : 0.0f;
    out[idx] = (half_t)v;
}

__global__ void zero_accum_kernel(float* accum) {
    if (threadIdx.x < NEXP) accum[threadIdx.x] = 0.0f;
}

__global__ void aux_loss_kernel(const float* __restrict__ accum, float* __restrict__ out_aux) {
    if (threadIdx.x == 0) {
        float ent = 0.0f, l2 = 0.0f;
        for (int j = 0; j < NEXP; ++j) {
            float a = accum[j] / (float)NTOK;
            ent -= a * logf(a + 1e-8f);
            float d = a - (1.0f / NEXP);
            l2 += d * d;
        }
        l2 /= (float)NEXP;
        out_aux[0] = -ent * 0.01f + 0.01f * l2;
    }
}

__global__ __launch_bounds__(256) void moe_main_kernel(
    const float* __restrict__ x,
    const float* __restrict__ rb1, const float* __restrict__ rb2, const float* __restrict__ rb3,
    const float* __restrict__ tb1, const float* __restrict__ tb2, const float* __restrict__ tb3,
    const float* __restrict__ sb1, const float* __restrict__ sb2, const float* __restrict__ sb3,
    const half_t* __restrict__ wsw,
    float* __restrict__ out_traj,   // [N,120]
    float* __restrict__ out_score,  // [N]
    float* __restrict__ out_probs,  // [N,6]
    float* __restrict__ probs_accum)
{
    __shared__ half_t xs[TM * 128];
    __shared__ half_t hA[TM * 256];
    __shared__ half_t hB[TM * 256];
    __shared__ float  logits_s[TM * 16];
    __shared__ float  topw[NEXP * TM];
    __shared__ float  psum[NEXP];

    const int tid  = threadIdx.x;
    const int wave = tid >> 5;
    const int lane = tid & 31;
    const int base_tok = blockIdx.x * TM;

    if (tid < NEXP) psum[tid] = 0.0f;
    for (int i = tid; i < TM * 128; i += 256)
        xs[i] = (half_t)x[(size_t)base_tok * 128 + i];
    __syncthreads();

    // ---------------- router ----------------
    gemm_lds<8, 16, 4, true>(xs, 128, wsw + OFS_RW1, rb1, 256, hA, 256, wave, lane);
    __syncthreads();
    gemm_lds<8, 8, 8, true>(hA, 256, wsw + OFS_RW2, rb2, 128, hB, 128, wave, lane);
    __syncthreads();
    {   // logits: [128,16] fp32 (kept in f32 through softmax/top-2: tie-sensitive)
        v16h breg[4];
#pragma unroll
        for (int k = 0; k < 4; ++k)
            breg[k] = load_b(wsw + OFS_RW3 + (size_t)k * 512, lane);
        v8f acc = {};
#pragma unroll
        for (int k = 0; k < 4; ++k) {
            v16h a = load_a(hB, 128, wave * 16, k * 32, lane);
            acc = wmma_f16(a, breg[k], acc);
        }
        int col = lane & 15;
        float bv = (col < NEXP) ? rb3[col] : 0.0f;
        int mrow = wave * 16 + ((lane >> 4) << 3);
#pragma unroll
        for (int v = 0; v < 8; ++v)
            logits_s[(mrow + v) * 16 + col] = acc[v] + bv;
    }
    __syncthreads();

    // ---------------- softmax + top-2 (one thread per token) ----------------
    if (tid < TM) {
        float lg[NEXP], mx = -1e30f;
        for (int j = 0; j < NEXP; ++j) { lg[j] = logits_s[tid * 16 + j]; mx = fmaxf(mx, lg[j]); }
        float s = 0.0f, p[NEXP];
        for (int j = 0; j < NEXP; ++j) { p[j] = expf(lg[j] - mx); s += p[j]; }
        float inv = 1.0f / s;
        size_t g = (size_t)(base_tok + tid) * NEXP;
        for (int j = 0; j < NEXP; ++j) {
            float pr = p[j] * inv;
            out_probs[g + j] = pr;
            atomicAdd(&psum[j], pr);
            topw[j * TM + tid] = 0.0f;
        }
        int i0 = 0;
        for (int j = 1; j < NEXP; ++j) if (lg[j] > lg[i0]) i0 = j;     // first-index on ties
        int i1 = (i0 == 0) ? 1 : 0;
        for (int j = 0; j < NEXP; ++j) if (j != i0 && lg[j] > lg[i1]) i1 = j;
        float w0 = 1.0f / (1.0f + expf(lg[i1] - lg[i0]));
        topw[i0 * TM + tid] = w0;
        topw[i1 * TM + tid] = 1.0f - w0;
    }
    __syncthreads();
    if (tid < NEXP) atomicAdd(&probs_accum[tid], psum[tid]);

    // ---------------- experts: fused heads, weighted accumulation in regs ----------------
    v8f tracc[8] = {};   // m-tiles 0..7 at column-tile n = wave
    v8f scacc = {};      // m-tile = wave, col 0

    for (int e = 0; e < NEXP; ++e) {
        __syncthreads();
        gemm_lds<8, 16, 4, true>(xs, 128, wsw + OFS_TW1 + (size_t)e * 32768,
                                 tb1 + e * 256, 256, hA, 256, wave, lane);
        __syncthreads();
        gemm_lds<8, 16, 8, true>(hA, 256, wsw + OFS_TW2 + (size_t)e * 65536,
                                 tb2 + e * 256, 256, hB, 256, wave, lane);
        __syncthreads();
        // trajectory output: wave owns n = wave; B regs reused across all 8 m-tiles
        {
            v16h breg[8];
#pragma unroll
            for (int k = 0; k < 8; ++k)
                breg[k] = load_b(wsw + OFS_TW3 + (size_t)e * 32768 + (size_t)(k * 8 + wave) * 512, lane);
            int col = wave * 16 + (lane & 15);
            float bv = (col < OUTW) ? tb3[e * OUTW + col] : 0.0f;
#pragma unroll 2
            for (int m = 0; m < 8; ++m) {
                v8f acc = {};
#pragma unroll
                for (int k = 0; k < 8; ++k) {
                    v16h a = load_a(hB, 256, m * 16, k * 32, lane);
                    acc = wmma_f16(a, breg[k], acc);
                }
                int mrow = m * 16 + ((lane >> 4) << 3);
#pragma unroll
                for (int v = 0; v < 8; ++v) {
                    float w = topw[e * TM + mrow + v];
                    tracc[m][v] += w * (acc[v] + bv);
                }
            }
        }
        // score head (hA/hB reuse is safe: traj reads of hB complete before the sync below)
        gemm_lds<8, 8, 4, true>(xs, 128, wsw + OFS_SW1 + (size_t)e * 16384,
                                sb1 + e * 128, 128, hA, 128, wave, lane);
        __syncthreads();
        gemm_lds<8, 4, 4, true>(hA, 128, wsw + OFS_SW2 + (size_t)e * 8192,
                                sb2 + e * 64, 64, hB, 64, wave, lane);
        __syncthreads();
        {
            v8f acc = {};
#pragma unroll
            for (int k = 0; k < 2; ++k) {
                v16h a = load_a(hB, 64, wave * 16, k * 32, lane);
                v16h b = load_b(wsw + OFS_SW3 + (size_t)e * 1024 + (size_t)k * 512, lane);
                acc = wmma_f16(a, b, acc);
            }
            int col = lane & 15;
            float bv = (col == 0) ? sb3[e] : 0.0f;
            int mrow = wave * 16 + ((lane >> 4) << 3);
#pragma unroll
            for (int v = 0; v < 8; ++v) {
                float w = topw[e * TM + mrow + v];
                scacc[v] += w * (acc[v] + bv);
            }
        }
    }

    // ---------------- write outputs ----------------
    {
        int col = wave * 16 + (lane & 15);
        if (col < OUTW) {
            for (int m = 0; m < 8; ++m) {
                int mrow = m * 16 + ((lane >> 4) << 3);
#pragma unroll
                for (int v = 0; v < 8; ++v)
                    out_traj[(size_t)(base_tok + mrow + v) * OUTW + col] = tracc[m][v];
            }
        }
    }
    if ((lane & 15) == 0) {
        int mrow = wave * 16 + ((lane >> 4) << 3);
#pragma unroll
        for (int v = 0; v < 8; ++v)
            out_score[base_tok + mrow + v] = scacc[v];
    }
}

extern "C" void kernel_launch(void* const* d_in, const int* in_sizes, int n_in,
                              void* d_out, int out_size, void* d_ws, size_t ws_size,
                              hipStream_t stream) {
    const float* x   = (const float*)d_in[0];
    const float* rW1 = (const float*)d_in[1];  const float* rb1 = (const float*)d_in[2];
    const float* rW2 = (const float*)d_in[3];  const float* rb2 = (const float*)d_in[4];
    const float* rW3 = (const float*)d_in[5];  const float* rb3 = (const float*)d_in[6];
    const float* tW1 = (const float*)d_in[7];  const float* tb1 = (const float*)d_in[8];
    const float* tW2 = (const float*)d_in[9];  const float* tb2 = (const float*)d_in[10];
    const float* tW3 = (const float*)d_in[11]; const float* tb3 = (const float*)d_in[12];
    const float* sW1 = (const float*)d_in[13]; const float* sb1 = (const float*)d_in[14];
    const float* sW2 = (const float*)d_in[15]; const float* sb2 = (const float*)d_in[16];
    const float* sW3 = (const float*)d_in[17]; const float* sb3 = (const float*)d_in[18];

    float*  accum = (float*)d_ws;                       // 6 floats
    half_t* wsw   = (half_t*)((char*)d_ws + 64);        // swizzled f16 weights (~2 MB)

    float* out       = (float*)d_out;
    float* out_traj  = out;                             // 49152*120
    float* out_score = out + (size_t)NTOK * OUTW;       // 49152
    float* out_aux   = out_score + NTOK;                // 1
    float* out_probs = out_aux + 1;                     // 49152*6

    zero_accum_kernel<<<1, 64, 0, stream>>>(accum);

    auto pack = [&](const float* W, size_t ofs, int K, int Nsrc, int Npad, int nmat) {
        long total = (long)(K / 32) * (Npad / 16) * 512 * nmat;
        int grid = (int)((total + 255) / 256);
        pack_b_kernel<<<grid, 256, 0, stream>>>(W, wsw + ofs, K, Nsrc, Npad, total);
    };
    pack(rW1, OFS_RW1, 128, 256, 256, 1);
    pack(rW2, OFS_RW2, 256, 128, 128, 1);
    pack(rW3, OFS_RW3, 128,   6,  16, 1);
    pack(tW1, OFS_TW1, 128, 256, 256, 6);
    pack(tW2, OFS_TW2, 256, 256, 256, 6);
    pack(tW3, OFS_TW3, 256, 120, 128, 6);
    pack(sW1, OFS_SW1, 128, 128, 128, 6);
    pack(sW2, OFS_SW2, 128,  64,  64, 6);
    pack(sW3, OFS_SW3,  64,   1,  16, 6);

    moe_main_kernel<<<NTOK / TM, 256, 0, stream>>>(
        x, rb1, rb2, rb3, tb1, tb2, tb3, sb1, sb2, sb3,
        wsw, out_traj, out_score, out_probs, accum);

    aux_loss_kernel<<<1, 32, 0, stream>>>(accum, out_aux);
}
